// CapsAll_51814485459029
// MI455X (gfx1250) — matrix operand order repaired and checked by reading, not compiled, not run on hardware
//
#include <hip/hip_runtime.h>
#include <hip/hip_bf16.h>
#include <stdint.h>

typedef __bf16 bf16;
typedef __attribute__((ext_vector_type(16))) __bf16 v16bf;
typedef __attribute__((ext_vector_type(8)))  float  v8f;
typedef __attribute__((ext_vector_type(4)))  unsigned int u32x4;
typedef __attribute__((ext_vector_type(8)))  int    i32x8;
typedef __attribute__((ext_vector_type(4)))  int    i32x4;

#define R_   8
#define U2_  1024
#define DA_  512
#define SC_  128
#define OA_  16
#define NR_  3
#define B_   128
#define T_   256
#define OTOT (SC_*OA_)   // 2048

// ---------------------------------------------------------------- WMMA helpers
__device__ __forceinline__ v8f v8f_zero() {
  v8f z = {0.f,0.f,0.f,0.f,0.f,0.f,0.f,0.f};
  return z;
}

__device__ __forceinline__ v8f wmma_bf16(v16bf a, v16bf b, v8f c) {
  // D = A(16x32 bf16) * B(32x16 bf16) + C(16x16 f32)
  return __builtin_amdgcn_wmma_f32_16x16x32_bf16(false, a, false, b, (short)0, c, false, false);
}

// A fragment: 16x32 bf16 tile, row-major source with leading dim `ld` (elements).
// lane 0..15 -> row M=lane, K 0..15; lane 16..31 -> row M=lane-16, K 16..31
__device__ __forceinline__ v16bf load_a_frag(const bf16* base, int ld, int row0, int k0, int lane) {
  const bf16* p = base + (size_t)(row0 + (lane & 15)) * ld + k0 + ((lane >> 4) << 4);
  union { v16bf v; uint4 q[2]; } u;
  u.q[0] = *(const uint4*)(p);
  u.q[1] = *(const uint4*)(p + 8);
  return u.v;
}

// B fragment: 32x16 bf16 tile; source stored as [n][K] (K contiguous), leading dim `ld`.
// lane 0..15 -> col N=lane, K 0..15; lane 16..31 -> col N=lane-16, K 16..31
__device__ __forceinline__ v16bf load_b_frag(const bf16* base, int ld, int col0, int k0, int lane) {
  const bf16* p = base + (size_t)(col0 + (lane & 15)) * ld + k0 + ((lane >> 4) << 4);
  union { v16bf v; uint4 q[2]; } u;
  u.q[0] = *(const uint4*)(p);
  u.q[1] = *(const uint4*)(p + 8);
  return u.v;
}

// ---------------------------------------------------------------- TDM (async tensor DMA)
#if __has_builtin(__builtin_amdgcn_tensor_load_to_lds) && __has_builtin(__builtin_amdgcn_s_wait_tensorcnt)
#define HAVE_TDM 1
#else
#define HAVE_TDM 0
#endif

#if HAVE_TDM
// 2D bf16 tile load: tensor [tensor_h][tensor_w] elements (row stride stride_w elems),
// tile [tile_h][tile_w] starting at gsrc, packed contiguously at LDS byte address lds_byte.
__device__ __forceinline__ void tdm_load_2d_bf16(const bf16* gsrc, uint32_t lds_byte,
                                                 uint32_t tensor_w, uint32_t tensor_h,
                                                 uint32_t stride_w, uint32_t tile_w,
                                                 uint32_t tile_h) {
  uint64_t ga = (uint64_t)(uintptr_t)gsrc;
  u32x4 g0 = {0u,0u,0u,0u};
  g0[0] = 1u;                                           // count=1, user descriptor
  g0[1] = lds_byte;                                     // lds_addr
  g0[2] = (uint32_t)ga;                                 // global_addr[31:0]
  g0[3] = (uint32_t)((ga >> 32) & 0x01FFFFFFu) | (2u << 30);  // addr hi + type=2
  i32x8 g1 = {0,0,0,0,0,0,0,0};
  g1[0] = (int)(1u << 16);                              // data_size = 2 bytes, mask=0
  g1[1] = (int)((tensor_w & 0xFFFFu) << 16);            // tensor_dim0[15:0]
  g1[2] = (int)(((tensor_w >> 16) & 0xFFFFu) | ((tensor_h & 0xFFFFu) << 16));
  g1[3] = (int)(((tensor_h >> 16) & 0xFFFFu) | ((tile_w & 0xFFFFu) << 16));
  g1[4] = (int)(tile_h & 0xFFFFu);                      // tile_dim1 (tile_dim2 = 0)
  g1[5] = (int)stride_w;                                // tensor_dim0_stride[31:0]
  g1[6] = 0;
  g1[7] = 0;
  i32x4 g2 = {0,0,0,0};
  i32x4 g3 = {0,0,0,0};
#if __clang_major__ >= 23
  i32x8 g4 = {0,0,0,0,0,0,0,0};
  __builtin_amdgcn_tensor_load_to_lds(g0, g1, g2, g3, g4, 0);
#else
  __builtin_amdgcn_tensor_load_to_lds(g0, g1, g2, g3, 0);
#endif
}
#endif

// ---------------------------------------------------------------- conversion kernels
__global__ void cvt_bf16_kernel(const float* __restrict__ s, bf16* __restrict__ d, int n) {
  for (int i = blockIdx.x * blockDim.x + threadIdx.x; i < n; i += gridDim.x * blockDim.x)
    d[i] = (bf16)s[i];
}

// caps [R][U2][2048] f32 -> capsT [R][2048][U2] bf16 (K=u contiguous for B-frags)
__global__ void capsT_kernel(const float* __restrict__ caps, bf16* __restrict__ ct) {
  int i = blockIdx.x * blockDim.x + threadIdx.x;
  if (i >= R_ * OTOT * U2_) return;
  int u = i & (U2_ - 1);
  int o = (i >> 10) & (OTOT - 1);
  int r = i >> 21;
  ct[i] = (bf16)caps[((size_t)r * U2_ + u) * OTOT + o];
}

// ---------------------------------------------------------------- fused attention kernel
// One WG per (b,r).  Phase A: hbar=relu(x_b@WS1_r^T) -> LDS (256KB).
// Phase B: scores in 64-col u-blocks via WMMA, softmax over T, m[b,r,u].
__launch_bounds__(256, 1)
__global__ void attn_m_kernel(const bf16* __restrict__ xbf,
                              const bf16* __restrict__ w1bf,
                              const bf16* __restrict__ w2bf,
                              bf16* __restrict__ mbf) {
  __shared__ __align__(16) bf16 hbar[T_ * DA_];            // 256KB  [256][512]
  __shared__ __align__(16) bf16 stage[(DA_ + 32) * 32];    // 34KB: wtile[512][32] + xtile[32][32]
  __shared__ float red[4][64];
  __shared__ float colstat[64];

  const int b    = blockIdx.x >> 3;
  const int r    = blockIdx.x & 7;
  const int tid  = threadIdx.x;
  const int lane = tid & 31;
  const int wave = tid >> 5;
  const int hi   = (lane >> 4) << 3;

  const bf16* xb  = xbf  + (size_t)b * T_ * U2_;
  const bf16* w1r = w1bf + (size_t)r * DA_ * U2_;
  const bf16* w2r = w2bf + (size_t)r * U2_ * DA_;

  bf16* wtile = stage;               // [512][32]
  bf16* xtile = stage + DA_ * 32;    // [32][32]

  // ---------------- Phase A ----------------
  for (int strip = 0; strip < 8; ++strip) {   // 32 T-rows per strip
    const int trow0 = strip * 32;
    v8f acc[2][4];
#pragma unroll
    for (int i = 0; i < 2; ++i)
#pragma unroll
      for (int j = 0; j < 4; ++j) acc[i][j] = v8f_zero();

    for (int kc = 0; kc < U2_; kc += 32) {
#if HAVE_TDM
      if (wave == 0)   // wave-level TDM: WS1_r strip [512 rows x 32 K] -> wtile
        tdm_load_2d_bf16(w1r + kc, (uint32_t)(uintptr_t)wtile,
                         U2_, DA_, U2_, 32, DA_);
#else
      for (int i = tid; i < (DA_ * 32) / 8; i += 256) {    // 2048 x 16B chunks
        int row = i >> 2, ch = i & 3;
        *(uint4*)(wtile + row * 32 + ch * 8) =
            *(const uint4*)(w1r + (size_t)row * U2_ + kc + ch * 8);
      }
#endif
      if (tid < 128) {                                     // xtile [32][32]
        int row = tid >> 2, ch = tid & 3;
        *(uint4*)(xtile + row * 32 + ch * 8) =
            *(const uint4*)(xb + (size_t)(trow0 + row) * U2_ + kc + ch * 8);
      }
#if HAVE_TDM
      if (wave == 0) __builtin_amdgcn_s_wait_tensorcnt(0);
#endif
      __syncthreads();

      v16bf af0 = load_a_frag(xtile, 32, 0, 0, lane);
      v16bf af1 = load_a_frag(xtile, 32, 16, 0, lane);
#pragma unroll
      for (int cb = 0; cb < 4; ++cb) {
        v16bf bf_ = load_b_frag(wtile, 32, (wave * 4 + cb) * 16, 0, lane);
        acc[0][cb] = wmma_bf16(af0, bf_, acc[0][cb]);
        acc[1][cb] = wmma_bf16(af1, bf_, acc[1][cb]);
      }
      __syncthreads();   // stage reused next kc
    }

    // relu + store strip into hbar (bf16)
#pragma unroll
    for (int rb = 0; rb < 2; ++rb)
#pragma unroll
      for (int cb = 0; cb < 4; ++cb) {
        int col   = (wave * 4 + cb) * 16 + (lane & 15);
        int rbase = trow0 + rb * 16 + hi;
#pragma unroll
        for (int v = 0; v < 8; ++v) {
          float f = acc[rb][cb][v];
          hbar[(size_t)(rbase + v) * DA_ + col] = (bf16)(f > 0.f ? f : 0.f);
        }
      }
  }
  __syncthreads();

  // ---------------- Phase B ----------------
  bf16* sc = stage;   // scores bf16 [256][64] (32KB, fits in stage)
  const int col = tid & 63, seg = tid >> 6;

  for (int ub = 0; ub < 16; ++ub) {
    const int u0 = ub * 64;
    v8f acc[2][4];
#pragma unroll
    for (int i = 0; i < 2; ++i)
#pragma unroll
      for (int j = 0; j < 4; ++j) acc[i][j] = v8f_zero();

    const int trow0 = wave * 32;       // this wave's 32 score rows
    for (int kc = 0; kc < DA_; kc += 32) {
      __builtin_prefetch(w2r + (size_t)(u0 + (lane & 63)) * DA_ + ((kc + 32) & (DA_ - 1)), 0, 1);
      v16bf af0 = load_a_frag(hbar, DA_, trow0, kc, lane);
      v16bf af1 = load_a_frag(hbar, DA_, trow0 + 16, kc, lane);
#pragma unroll
      for (int cb = 0; cb < 4; ++cb) {
        v16bf bf_ = load_b_frag(w2r, DA_, u0 + cb * 16, kc, lane);   // direct global
        acc[0][cb] = wmma_bf16(af0, bf_, acc[0][cb]);
        acc[1][cb] = wmma_bf16(af1, bf_, acc[1][cb]);
      }
    }
    __syncthreads();   // previous ublock's softmax reads of sc are done

#pragma unroll
    for (int rb = 0; rb < 2; ++rb)
#pragma unroll
      for (int cb = 0; cb < 4; ++cb)
#pragma unroll
        for (int v = 0; v < 8; ++v)
          sc[(size_t)(trow0 + rb * 16 + hi + v) * 64 + cb * 16 + (lane & 15)] =
              (bf16)acc[rb][cb][v];
    __syncthreads();

    // softmax over T (256 rows) per column, 4 thread-segments per column
    float mx = -3.0e38f;
    for (int t = seg * 64; t < seg * 64 + 64; ++t)
      mx = fmaxf(mx, (float)sc[(size_t)t * 64 + col]);
    red[seg][col] = mx;
    __syncthreads();
    if (seg == 0)
      colstat[col] = fmaxf(fmaxf(red[0][col], red[1][col]), fmaxf(red[2][col], red[3][col]));
    __syncthreads();
    const float M = colstat[col];

    float s = 0.f;
    for (int t = seg * 64; t < seg * 64 + 64; ++t)
      s += __expf((float)sc[(size_t)t * 64 + col] - M);
    red[seg][col] = s;
    __syncthreads();
    if (seg == 0)
      colstat[col] = red[0][col] + red[1][col] + red[2][col] + red[3][col];
    __syncthreads();
    const float S = colstat[col];

    // m[u] = sum_t softmax(score)[t,u] * x[b,t,u]
    float am = 0.f;
    for (int t = seg * 64; t < seg * 64 + 64; ++t)
      am += __expf((float)sc[(size_t)t * 64 + col] - M) *
            (float)xb[(size_t)t * U2_ + u0 + col];
    red[seg][col] = am;
    __syncthreads();
    if (seg == 0) {
      float mval = (red[0][col] + red[1][col] + red[2][col] + red[3][col]) / S;
      mbf[((size_t)r * B_ + b) * U2_ + u0 + col] = (bf16)mval;
    }
    __syncthreads();
  }
}

// ---------------------------------------------------------------- votes GEMM
// per r: votes = m_r[128x1024] @ capsT_r^T  -> [128 x 2048]; WG = (oblk, r)
__launch_bounds__(256, 1)
__global__ void votes_kernel(const bf16* __restrict__ mbf, const bf16* __restrict__ capsT,
                             float* __restrict__ votes) {
  const int ob   = blockIdx.x;          // 16 blocks of 128 output cols
  const int r    = blockIdx.y;          // 8
  const int lane = threadIdx.x & 31;
  const int wave = threadIdx.x >> 5;
  const int hi   = (lane >> 4) << 3;
  const bf16* A  = mbf   + (size_t)r * B_ * U2_;      // [128][1024]
  const bf16* Bm = capsT + (size_t)r * OTOT * U2_;    // [2048][1024]

  v8f acc[8];
#pragma unroll
  for (int i = 0; i < 8; ++i) acc[i] = v8f_zero();

  for (int kc = 0; kc < U2_; kc += 32) {
    v16bf af = load_a_frag(A, U2_, wave * 16, kc, lane);
#pragma unroll
    for (int cb = 0; cb < 8; ++cb) {
      v16bf bf_ = load_b_frag(Bm, U2_, ob * 128 + cb * 16, kc, lane);
      acc[cb] = wmma_bf16(af, bf_, acc[cb]);
    }
  }
#pragma unroll
  for (int cb = 0; cb < 8; ++cb) {
    int op = ob * 128 + cb * 16 + (lane & 15);
    int c = op >> 4, o = op & 15;
#pragma unroll
    for (int v = 0; v < 8; ++v) {
      int bb = wave * 16 + hi + v;
      votes[(((size_t)bb * R_ + r) * SC_ + c) * OA_ + o] = acc[cb][v];
    }
  }
}

// ---------------------------------------------------------------- dynamic routing
__launch_bounds__(128, 1)
__global__ void routing_kernel(const float* __restrict__ votes, float* __restrict__ out) {
  __shared__ float vsh[R_][SC_][OA_];   // 64KB
  __shared__ float logit[R_][SC_];
  __shared__ float act[SC_][OA_];
  __shared__ float rmax[R_], rsum[R_];
  const int b = blockIdx.x, c = threadIdx.x;

  for (int i = threadIdx.x; i < R_ * SC_ * OA_; i += 128)
    ((float*)vsh)[i] = votes[(size_t)b * R_ * SC_ * OA_ + i];
  for (int rr = 0; rr < R_; ++rr) logit[rr][c] = 0.f;
  __syncthreads();

  for (int it = 0; it < NR_; ++it) {
    if (c < R_) {     // per-r softmax stats over SC
      float mx = -3.0e38f;
      for (int cc = 0; cc < SC_; ++cc) mx = fmaxf(mx, logit[c][cc]);
      float s = 0.f;
      for (int cc = 0; cc < SC_; ++cc) s += __expf(logit[c][cc] - mx);
      rmax[c] = mx; rsum[c] = s;
    }
    __syncthreads();
    float pre[OA_];
#pragma unroll
    for (int o = 0; o < OA_; ++o) pre[o] = 0.f;
    for (int rr = 0; rr < R_; ++rr) {
      float route = __expf(logit[rr][c] - rmax[rr]) / rsum[rr];
#pragma unroll
      for (int o = 0; o < OA_; ++o) pre[o] += route * vsh[rr][c][o];
    }
    float n2 = 0.f;
#pragma unroll
    for (int o = 0; o < OA_; ++o) n2 += pre[o] * pre[o];
    float scale = n2 / ((1.f + n2) * sqrtf(n2));
#pragma unroll
    for (int o = 0; o < OA_; ++o) act[c][o] = pre[o] * scale;
    __syncthreads();
    for (int rr = 0; rr < R_; ++rr) {
      float d = 0.f;
#pragma unroll
      for (int o = 0; o < OA_; ++o) d += vsh[rr][c][o] * act[c][o];
      logit[rr][c] += d;
    }
    __syncthreads();
  }
  float n2 = 0.f;
#pragma unroll
  for (int o = 0; o < OA_; ++o) { float a = act[c][o]; n2 += a * a; }
  out[(size_t)b * SC_ + c] = sqrtf(n2);
}

// ---------------------------------------------------------------- host launcher
extern "C" void kernel_launch(void* const* d_in, const int* in_sizes, int n_in,
                              void* d_out, int out_size, void* d_ws, size_t ws_size,
                              hipStream_t stream) {
  (void)in_sizes; (void)n_in; (void)out_size; (void)ws_size;
  const float* x    = (const float*)d_in[0];   // [B,T,U2]
  const float* WS1  = (const float*)d_in[1];   // [R,DA,U2]
  const float* WS2  = (const float*)d_in[2];   // [R,U2,DA]
  const float* caps = (const float*)d_in[3];   // [R,U2,SC*OA]

  char* ws = (char*)d_ws;
  const size_t NX  = (size_t)B_ * T_ * U2_;     // 33,554,432
  const size_t NW1 = (size_t)R_ * DA_ * U2_;    //  4,194,304
  const size_t NW2 = (size_t)R_ * U2_ * DA_;    //  4,194,304
  const size_t NCT = (size_t)R_ * OTOT * U2_;   // 16,777,216
  size_t off = 0;
  bf16*  xbf  = (bf16*)(ws + off); off += NX  * 2;
  bf16*  w1bf = (bf16*)(ws + off); off += NW1 * 2;
  bf16*  w2bf = (bf16*)(ws + off); off += NW2 * 2;
  bf16*  ct   = (bf16*)(ws + off); off += NCT * 2;
  bf16*  mbf  = (bf16*)(ws + off); off += (size_t)R_ * B_ * U2_ * 2;
  float* vts  = (float*)(ws + off);            // [B][R][SC][OA] f32

  cvt_bf16_kernel<<<8192, 256, 0, stream>>>(x,   xbf,  (int)NX);
  cvt_bf16_kernel<<<4096, 256, 0, stream>>>(WS1, w1bf, (int)NW1);
  cvt_bf16_kernel<<<4096, 256, 0, stream>>>(WS2, w2bf, (int)NW2);
  capsT_kernel<<<(int)(NCT / 256), 256, 0, stream>>>(caps, ct);

  attn_m_kernel<<<B_ * R_, 256, 0, stream>>>(xbf, w1bf, w2bf, mbf);
  votes_kernel<<<dim3(16, 8), 256, 0, stream>>>(mbf, ct, vts);
  routing_kernel<<<B_, SC_, 0, stream>>>(vts, (float*)d_out);
}
